// MultiHeadAttention_79328045957555
// MI455X (gfx1250) — matrix operand order repaired and checked by reading, not compiled, not run on hardware
//
#include <hip/hip_runtime.h>

// ---------------------------------------------------------------------------
// MHA forward for MI455X (gfx1250, wave32, WMMA 16x16x32 f16 -> f32 acc)
// B=2, S=2048, D=1024, H=16, Dh=64
// ---------------------------------------------------------------------------

#define B_  2
#define S_  2048
#define D_  1024
#define H_  16
#define DH_ 64
#define M_ROWS (B_ * S_)   // 4096
#define N_QKV  (3 * D_)    // 3072

typedef __attribute__((ext_vector_type(16))) _Float16 v16h;
typedef __attribute__((ext_vector_type(8)))  _Float16 v8h;
typedef __attribute__((ext_vector_type(8)))  float    v8f;

// A-matrix (16x32 f16) per-lane load, contiguous k order (matches B loads):
// lane L: m = L%16, base = (L>=16 ? 8 : 0)
//   halves 0..7  = A[m][2*base + j]
//   halves 8..15 = A[m][2*base + 8 + j]
__device__ __forceinline__ v16h load_a16(const _Float16* row, int base) {
  v8h lo = *(const v8h*)(row + 2 * base);
  v8h hh = *(const v8h*)(row + 2 * base + 8);
  return __builtin_shufflevector(lo, hh, 0,1,2,3,4,5,6,7,8,9,10,11,12,13,14,15);
}

__device__ __forceinline__ v8f wmma_f16(v16h a, v16h b, v8f c) {
  // D = A(16x32) * B(32x16) + C, fp32 accumulate
  return __builtin_amdgcn_wmma_f32_16x16x32_f16(false, a, false, b, (short)0, c,
                                                false, false);
}

// ---------------------------------------------------------------------------
// Kernel 0: fp32 -> fp16 convert (grid-stride)
// ---------------------------------------------------------------------------
__global__ void cvt_f32_f16(const float* __restrict__ in,
                            _Float16* __restrict__ out, int n) {
  for (int i = blockIdx.x * blockDim.x + threadIdx.x; i < n;
       i += gridDim.x * blockDim.x) {
    out[i] = (_Float16)in[i];
  }
}

// ---------------------------------------------------------------------------
// Kernel 1/4: C[M,N] = A[M,K] * W[N,K]^T + bias,  f16 inputs, f32 accumulate.
// Block = 256 threads (8 waves). Wave tile = 32 rows x 64 cols: two A row
// tiles share every B tile (8 WMMAs per 32-K chunk, ~21 FLOP per L2 byte).
// Workgroup tile = 64 rows x 256 cols.
// ---------------------------------------------------------------------------
template <bool OUT_F16>
__global__ __launch_bounds__(256) void gemm_bias(
    const _Float16* __restrict__ A, const _Float16* __restrict__ W,
    const float* __restrict__ bias, void* __restrict__ Cout,
    int M, int N, int K) {
  const int lane = threadIdx.x & 31;
  const int w    = threadIdx.x >> 5;
  const int lm   = lane & 15;          // column / row-within-tile index
  const int hig  = lane >> 4;          // half-wave group

  const int m0 = blockIdx.x * 64 + (w & 1) * 32;
  const int n0 = blockIdx.y * 256 + (w >> 1) * 64;

  v8f acc[2][4];
#pragma unroll
  for (int rt = 0; rt < 2; ++rt)
#pragma unroll
    for (int t = 0; t < 4; ++t) acc[rt][t] = (v8f){};

  const _Float16* arow0 = A + (size_t)(m0 + lm) * K;
  const _Float16* arow1 = A + (size_t)(m0 + 16 + lm) * K;
  const int abase = hig * 8;
  const int koff  = hig * 16;

  for (int k0 = 0; k0 < K; k0 += 32) {
    __builtin_prefetch(arow0 + k0 + 256, 0, 0);  // global_prefetch_b8
    v16h a0 = load_a16(arow0 + k0, abase);
    v16h a1 = load_a16(arow1 + k0, abase);
#pragma unroll
    for (int t = 0; t < 4; ++t) {
      // B[kk][n] = W[n0 + t*16 + n][k0 + kk]; lane=n holds 16 contiguous K vals
      const _Float16* wrow = W + (size_t)(n0 + t * 16 + lm) * K + k0 + koff;
      v16h b = *(const v16h*)wrow;
      acc[0][t] = wmma_f16(a0, b, acc[0][t]);
      acc[1][t] = wmma_f16(a1, b, acc[1][t]);
    }
  }

#pragma unroll
  for (int rt = 0; rt < 2; ++rt) {
#pragma unroll
    for (int t = 0; t < 4; ++t) {
      const int col = n0 + t * 16 + lm;
      const float bv = bias[col];
#pragma unroll
      for (int r = 0; r < 8; ++r) {
        const int row = m0 + rt * 16 + r + hig * 8;
        const float val = acc[rt][t][r] + bv;
        if (OUT_F16)
          ((_Float16*)Cout)[(size_t)row * N + col] = (_Float16)val;
        else
          ((float*)Cout)[(size_t)row * N + col] = val;
      }
    }
  }
}

// ---------------------------------------------------------------------------
// Kernel 2: RoPE on q,k + scatter to per-head layouts; V transposed to
// Vt[b][h][dh][s] so P*V WMMA B-operands are contiguous along s.
// One thread per (b,h,s,pair i), i in [0,32).
// ---------------------------------------------------------------------------
__global__ void rope_scatter(const _Float16* __restrict__ qkv,
                             _Float16* __restrict__ Qh,
                             _Float16* __restrict__ Kh,
                             _Float16* __restrict__ Vt) {
  const int tid = blockIdx.x * blockDim.x + threadIdx.x;
  const int i = tid & 31;
  const int s = (tid >> 5) & (S_ - 1);
  const int h = (tid >> 16) & (H_ - 1);          // 32*2048 = 2^16
  const int b = tid >> 20;
  if (b >= B_) return;

  const size_t row = (size_t)(b * S_ + s) * N_QKV;
  const int col = h * DH_ + 2 * i;
  const float q1 = (float)qkv[row + col];
  const float q2 = (float)qkv[row + col + 1];
  const float k1 = (float)qkv[row + D_ + col];
  const float k2 = (float)qkv[row + D_ + col + 1];
  const float v1 = (float)qkv[row + 2 * D_ + col];
  const float v2 = (float)qkv[row + 2 * D_ + col + 1];

  // freq_i = 10000^(-2i/64);  theta = s * freq_i
  const float inv_freq = __expf(-(float)(2 * i) * (9.210340371976184f / 64.0f));
  const float theta = (float)s * inv_freq;
  const float c = cosf(theta), sn = sinf(theta);

  const size_t bh = (size_t)(b * H_ + h);
  const size_t qk_base = (bh * S_ + s) * DH_ + 2 * i;
  Qh[qk_base]     = (_Float16)(q1 * c - q2 * sn);
  Qh[qk_base + 1] = (_Float16)(q1 * sn + q2 * c);
  Kh[qk_base]     = (_Float16)(k1 * c - k2 * sn);
  Kh[qk_base + 1] = (_Float16)(k1 * sn + k2 * c);

  Vt[(bh * DH_ + 2 * i) * S_ + s]     = (_Float16)v1;
  Vt[(bh * DH_ + 2 * i + 1) * S_ + s] = (_Float16)v2;
}

// ---------------------------------------------------------------------------
// Kernel 3: causal flash attention. Block = 256 thr (8 waves); each wave owns
// 16 query rows of one (b,h). Online softmax; scores & PV via WMMA; P tile is
// round-tripped through LDS to convert C-layout -> A-layout.
// ---------------------------------------------------------------------------
__global__ __launch_bounds__(256) void flash_attn(
    const _Float16* __restrict__ Qh, const _Float16* __restrict__ Kh,
    const _Float16* __restrict__ Vt, _Float16* __restrict__ Oh) {
  __shared__ _Float16 Plds[8 * 16 * 32];  // per-wave 16x32 f16 P tile

  const int lane = threadIdx.x & 31;
  const int w    = threadIdx.x >> 5;
  const int lm   = lane & 15;
  const int hig  = lane >> 4;

  const int qt = blockIdx.x % (S_ / 128);
  const int h  = (blockIdx.x / (S_ / 128)) % H_;
  const int b  = blockIdx.x / ((S_ / 128) * H_);
  const int qbase = qt * 128 + w * 16;

  const size_t bh = (size_t)(b * H_ + h);
  const _Float16* Qb = Qh + (bh * S_) * DH_;
  const _Float16* Kb = Kh + (bh * S_) * DH_;
  const _Float16* Vb = Vt + (bh * DH_) * S_;
  _Float16* Pw = Plds + w * (16 * 32);

  // Q rows in A-layout: two K-chunks of 32 covering Dh=64
  const _Float16* qrow = Qb + (size_t)(qbase + lm) * DH_;
  const int abase = hig * 8;
  const v16h aq0 = load_a16(qrow, abase);
  const v16h aq1 = load_a16(qrow + 32, abase);

  v8f acc[4];
#pragma unroll
  for (int t = 0; t < 4; ++t) acc[t] = (v8f){};
  float mrow[8], lrow[8];
#pragma unroll
  for (int r = 0; r < 8; ++r) { mrow[r] = -3.0e38f; lrow[r] = 0.0f; }

  const int nkb = (qbase + 16 + 31) / 32;  // causal: key blocks of 32
  for (int kb = 0; kb < nkb; ++kb) {
    const int kbase = kb * 32;

    // ---- scores: two 16x16 tiles (keys kbase..+15, kbase+16..+31) ----
    v8f s0 = (v8f){}, s1 = (v8f){};
    {
      const _Float16* k0row = Kb + (size_t)(kbase + lm) * DH_ + hig * 16;
      const _Float16* k1row = Kb + (size_t)(kbase + 16 + lm) * DH_ + hig * 16;
      v16h b00 = *(const v16h*)(k0row);
      v16h b01 = *(const v16h*)(k0row + 32);
      v16h b10 = *(const v16h*)(k1row);
      v16h b11 = *(const v16h*)(k1row + 32);
      s0 = wmma_f16(aq0, b00, s0);
      s0 = wmma_f16(aq1, b01, s0);
      s1 = wmma_f16(aq0, b10, s1);
      s1 = wmma_f16(aq1, b11, s1);
    }

    // ---- scale + causal mask + online softmax ----
    float p0[8], p1[8];
#pragma unroll
    for (int r = 0; r < 8; ++r) {
      const int row = qbase + r + hig * 8;
      float v0 = s0[r] * 0.125f;             // 1/sqrt(64)
      float v1 = s1[r] * 0.125f;
      if (kbase + lm > row)      v0 = -3.0e38f;
      if (kbase + 16 + lm > row) v1 = -3.0e38f;

      float bm = fmaxf(v0, v1);
      bm = fmaxf(bm, __shfl_xor(bm, 1, 16));
      bm = fmaxf(bm, __shfl_xor(bm, 2, 16));
      bm = fmaxf(bm, __shfl_xor(bm, 4, 16));
      bm = fmaxf(bm, __shfl_xor(bm, 8, 16));

      const float mnew  = fmaxf(mrow[r], bm);
      const float alpha = __expf(mrow[r] - mnew);
      p0[r] = __expf(v0 - mnew);
      p1[r] = __expf(v1 - mnew);

      float rs = p0[r] + p1[r];
      rs += __shfl_xor(rs, 1, 16);
      rs += __shfl_xor(rs, 2, 16);
      rs += __shfl_xor(rs, 4, 16);
      rs += __shfl_xor(rs, 8, 16);

      lrow[r] = lrow[r] * alpha + rs;
      mrow[r] = mnew;
#pragma unroll
      for (int t = 0; t < 4; ++t) acc[t][r] *= alpha;
    }

    // ---- stage P (C-layout) into LDS, re-read in A-layout ----
#pragma unroll
    for (int r = 0; r < 8; ++r) {
      const int prow = r + hig * 8;
      Pw[prow * 32 + lm]      = (_Float16)p0[r];
      Pw[prow * 32 + 16 + lm] = (_Float16)p1[r];
    }
    asm volatile("s_wait_dscnt 0" ::: "memory");  // DS store->load, same wave
    const v16h ap = load_a16(Pw + lm * 32, abase);

    // ---- O += P * V : 4 dh tiles, K=32 each ----
#pragma unroll
    for (int t = 0; t < 4; ++t) {
      const _Float16* vrow =
          Vb + (size_t)(t * 16 + lm) * S_ + kbase + hig * 16;
      v16h bv = *(const v16h*)vrow;
      acc[t] = wmma_f16(ap, bv, acc[t]);
    }
  }

  // ---- epilogue: normalize, write O as f16 in (B,S,H*Dh) layout ----
#pragma unroll
  for (int r = 0; r < 8; ++r) {
    const float inv_l = 1.0f / lrow[r];
    const int row = qbase + r + hig * 8;
#pragma unroll
    for (int t = 0; t < 4; ++t) {
      const int col = h * DH_ + t * 16 + lm;
      Oh[(size_t)(b * S_ + row) * D_ + col] = (_Float16)(acc[t][r] * inv_l);
    }
  }
}

// ---------------------------------------------------------------------------
// Launcher
// ---------------------------------------------------------------------------
extern "C" void kernel_launch(void* const* d_in, const int* in_sizes, int n_in,
                              void* d_out, int out_size, void* d_ws,
                              size_t ws_size, hipStream_t stream) {
  const float* x     = (const float*)d_in[0];   // (2, 2048, 1024)
  const float* w_qkv = (const float*)d_in[1];   // (3072, 1024)
  const float* b_qkv = (const float*)d_in[2];   // (3072,)
  const float* w_out = (const float*)d_in[3];   // (1024, 1024)
  const float* b_out = (const float*)d_in[4];   // (1024,)
  float* out = (float*)d_out;                   // (2, 2048, 1024)

  char* ws = (char*)d_ws;
  const size_t MB = 1024 * 1024;
  _Float16* xh   = (_Float16*)(ws + 0);        //  8 MB: x as f16
  _Float16* wqh  = (_Float16*)(ws + 8 * MB);   //  6 MB: w_qkv f16
  _Float16* woh  = (_Float16*)(ws + 14 * MB);  //  2 MB: w_out f16
  _Float16* qkvh = (_Float16*)(ws + 16 * MB);  // 24 MB: qkv f16 (bias fused)
  _Float16* Qh   = (_Float16*)(ws + 40 * MB);  //  8 MB: (b,h,s,dh) rope'd
  _Float16* Kh   = (_Float16*)(ws + 48 * MB);  //  8 MB: (b,h,s,dh) rope'd
  _Float16* Vt   = (_Float16*)(ws + 56 * MB);  //  8 MB: (b,h,dh,s)
  _Float16* oh   = (_Float16*)(ws + 64 * MB);  //  8 MB: attn out f16

  // 0) fp32 -> fp16 converts
  cvt_f32_f16<<<2048, 256, 0, stream>>>(x, xh, M_ROWS * D_);
  cvt_f32_f16<<<2048, 256, 0, stream>>>(w_qkv, wqh, N_QKV * D_);
  cvt_f32_f16<<<1024, 256, 0, stream>>>(w_out, woh, D_ * D_);

  // 1) qkv = x @ w_qkv^T + b_qkv   (M=4096, N=3072, K=1024), f16 out
  gemm_bias<true><<<dim3(M_ROWS / 64, N_QKV / 256), 256, 0, stream>>>(
      xh, wqh, b_qkv, (void*)qkvh, M_ROWS, N_QKV, D_);

  // 2) RoPE q,k; transpose v
  rope_scatter<<<(B_ * H_ * S_ * 32) / 256, 256, 0, stream>>>(qkvh, Qh, Kh, Vt);

  // 3) causal flash attention -> oh (B,S,D) f16
  flash_attn<<<B_ * H_ * (S_ / 128), 256, 0, stream>>>(Qh, Kh, Vt, oh);

  // 4) out = oh @ w_out^T + b_out  (M=4096, N=1024, K=1024), f32 out
  gemm_bias<false><<<dim3(M_ROWS / 64, D_ / 256), 256, 0, stream>>>(
      oh, woh, b_out, (void*)out, M_ROWS, D_, D_);
}
